// DQN_86861418594736
// MI455X (gfx1250) — matrix operand order repaired and checked by reading, not compiled, not run on hardware
//
#include <hip/hip_runtime.h>
#include <hip/hip_bf16.h>
#include <cstdint>

typedef __attribute__((ext_vector_type(16))) _Float16 v16h;
typedef __attribute__((ext_vector_type(8)))  float    v8f;
typedef __attribute__((ext_vector_type(4)))  float    f32x4;

union AF {
    v16h v;
    unsigned int   u[8];
    unsigned short s[16];
    f32x4          q[2];
};

// ---------------------------------------------------------------------------
// Stage 0: binarize + transpose  (B,75,200) -> (B,200,75) f16, {0,1}
// ---------------------------------------------------------------------------
__global__ __launch_bounds__(256) void binarize_transpose(
    const float* __restrict__ rgb, _Float16* __restrict__ out, int B)
{
    int t = blockIdx.x * 256 + threadIdx.x;
    const int total = B * 15000;
    if (t >= total) return;
    const int x = t % 75;
    const int y = (t / 75) % 200;
    const int b = t / 15000;
    const float v = rgb[b * 15000 + x * 200 + y];
    out[t] = (_Float16)(v > 0.f ? 1.f : 0.f);
}

// ---------------------------------------------------------------------------
// Weight packers: produce B-fragments in exact WMMA lane order.
// Layout: bp[((nt*KC + kb)*32 + lane)*16 + j], where
//   k = kb*32 + (j/8)*16 + (lane/16)*8 + j%8  (K = tap*CIN + c, taps row-major)
//   n = nt*16 + lane%16
// Out-of-range entries are zero (inert padding).
// ---------------------------------------------------------------------------
template<int CIN, int COUT, int KHW>
__global__ __launch_bounds__(256) void pack_conv_w(
    const float* __restrict__ w, _Float16* __restrict__ bp)
{
    constexpr int KT = CIN * KHW * KHW;
    constexpr int KC = (KT + 31) / 32;
    constexpr int NT = (COUT + 15) / 16;
    const int total = NT * KC * 512;
    int t = blockIdx.x * 256 + threadIdx.x;
    if (t >= total) return;
    const int j    = t & 15;
    const int lane = (t >> 4) & 31;
    const int kb   = (t >> 9) % KC;
    const int nt   = t / (KC * 512);
    const int k = kb * 32 + (j >> 3) * 16 + (lane >> 4) * 8 + (j & 7);
    const int n = nt * 16 + (lane & 15);
    float v = 0.f;
    if (n < COUT && k < KT) {
        const int tap = k / CIN, c = k - tap * CIN;
        const int ky = tap / KHW, kx = tap - ky * KHW;
        v = w[((n * CIN + c) * KHW + ky) * KHW + kx];   // OIHW
    }
    bp[t] = (_Float16)v;
}

__global__ __launch_bounds__(256) void pack_fc_w(
    const float* __restrict__ w, _Float16* __restrict__ bp, int K, int N)
{
    const int KC = (K + 31) / 32;
    const int NT = (N + 15) / 16;
    const int total = NT * KC * 512;
    int t = blockIdx.x * 256 + threadIdx.x;
    if (t >= total) return;
    const int j    = t & 15;
    const int lane = (t >> 4) & 31;
    const int kb   = (t >> 9) % KC;
    const int nt   = t / (KC * 512);
    const int k = kb * 32 + (j >> 3) * 16 + (lane >> 4) * 8 + (j & 7);
    const int n = nt * 16 + (lane & 15);
    bp[t] = (_Float16)((n < N && k < K) ? w[k * N + n] : 0.f);
}

// ---------------------------------------------------------------------------
// Implicit-GEMM convolution, NHWC f16 in, f16 out (NHWC or NCHW), ReLU+bias.
// One wave = 16 output pixels x ALL output channels: the (expensive) gathered
// A fragment is loaded once per k-chunk and reused for NT WMMAs against NT
// B-fragments read from LDS (whole packed weight set staged per block).
// ---------------------------------------------------------------------------
template<int CIN, int COUT, int KHW, int STRIDE, bool NCHW_OUT>
__global__ __launch_bounds__(256) void conv_wmma(
    const _Float16* __restrict__ in, const _Float16* __restrict__ wp,
    const float* __restrict__ bias, _Float16* __restrict__ out,
    int Hin, int Win, int Hout, int Wout, int Btot)
{
    constexpr int KT = CIN * KHW * KHW;
    constexpr int KC = (KT + 31) / 32;
    constexpr int NT = (COUT + 15) / 16;
    extern __shared__ _Float16 smB[];          // NT*KC*512 halves (dynamic)

    {   // cooperative stage of all packed weights for this layer
        const f32x4* src = (const f32x4*)wp;
        f32x4* dst = (f32x4*)smB;
        for (int i = threadIdx.x; i < NT * KC * 64; i += 256) dst[i] = src[i];
    }
    __syncthreads();

    const int lane = threadIdx.x & 31;
    const int wave = threadIdx.x >> 5;
    const int P  = Hout * Wout;
    const int MP = Btot * P;
    const int mt16 = (blockIdx.x * 8 + wave) * 16;
    if (mt16 >= MP) return;

    const int kgrp = lane >> 4;              // K half selector per A/B layout
    int g = mt16 + (lane & 15);
    if (g >= MP) g = MP - 1;
    const int b  = g / P;
    const int p  = g - b * P;
    const int ho = p / Wout;
    const int wo = p - ho * Wout;
    const int ibase = (b * Hin + ho * STRIDE) * Win + wo * STRIDE; // pixel idx

    v8f acc[NT];
    #pragma unroll
    for (int t = 0; t < NT; ++t) acc[t] = (v8f){0.f,0.f,0.f,0.f,0.f,0.f,0.f,0.f};

    for (int kb = 0; kb < KC; ++kb) {
        AF a;
        #pragma unroll
        for (int pr = 0; pr < 8; ++pr) {
            const int k = kb * 32 + (pr >> 2) * 16 + kgrp * 8 + (pr & 3) * 2;
            if constexpr (CIN == 1) {      // pairs span taps: u16 gathers
                unsigned short lo = 0, hi = 0;
                if (k < KT) {
                    int ky = k / KHW, kx = k - ky * KHW;
                    lo = *(const unsigned short*)(in + ibase + ky * Win + kx);
                }
                if (k + 1 < KT) {
                    int k1 = k + 1;
                    int ky = k1 / KHW, kx = k1 - ky * KHW;
                    hi = *(const unsigned short*)(in + ibase + ky * Win + kx);
                }
                a.s[2 * pr]     = lo;
                a.s[2 * pr + 1] = hi;
            } else {                       // even CIN: aligned contiguous pair
                unsigned int v = 0;
                if (k < KT) {
                    const int tap = k / CIN, c = k - tap * CIN;
                    const int ky = tap / KHW, kx = tap - ky * KHW;
                    v = *(const unsigned int*)(in + (ibase + ky * Win + kx) * CIN + c);
                }
                a.u[pr] = v;
            }
        }
        #pragma unroll
        for (int t = 0; t < NT; ++t) {     // reuse A across all n-tiles
            AF bb;
            const f32x4* bs = (const f32x4*)(smB + ((t * KC + kb) * 32 + lane) * 16);
            bb.q[0] = bs[0];
            bb.q[1] = bs[1];
            acc[t] = __builtin_amdgcn_wmma_f32_16x16x32_f16(
                false, a.v, false, bb.v, (short)0, acc[t], false, false);
        }
    }

    #pragma unroll
    for (int t = 0; t < NT; ++t) {
        const int n = t * 16 + (lane & 15);
        const float bz = (n < COUT) ? bias[n] : 0.f;
        #pragma unroll
        for (int r = 0; r < 8; ++r) {
            const int gs = mt16 + r + (kgrp << 3);
            if (gs < MP && n < COUT) {
                float v = acc[t][r] + bz;
                v = v > 0.f ? v : 0.f;
                if constexpr (NCHW_OUT) {
                    const int b2 = gs / P, p2 = gs - b2 * P;
                    out[(b2 * COUT + n) * P + p2] = (_Float16)v;
                } else {
                    out[(size_t)gs * COUT + n] = (_Float16)v;
                }
            }
        }
    }
}

// ---------------------------------------------------------------------------
// f16 GEMM + bias + ReLU, f32 out. Used for fc1 (M=B, K=5120, N=100).
// A rows are contiguous (K multiple of 32) -> 2x b128 loads/lane/chunk.
// ---------------------------------------------------------------------------
__global__ __launch_bounds__(256) void gemm_wmma_f16(
    const _Float16* __restrict__ A, const _Float16* __restrict__ wp,
    const float* __restrict__ bias, float* __restrict__ out,
    int M, int K, int N)
{
    const int lane = threadIdx.x & 31;
    const int wave = threadIdx.x >> 5;
    const int KB = K >> 5;
    const int nt = blockIdx.y;
    const int mt16 = (blockIdx.x * 8 + wave) * 16;
    if (mt16 >= M) return;
    const int kgrp = lane >> 4;
    int m = mt16 + (lane & 15);
    if (m >= M) m = M - 1;
    const _Float16* Ar = A + (size_t)m * K;
    const _Float16* Bp = wp + (size_t)nt * KB * 512;

    v8f acc = {0.f, 0.f, 0.f, 0.f, 0.f, 0.f, 0.f, 0.f};
    for (int kb = 0; kb < KB; ++kb) {
        AF a, bb;
        const int off = kb * 32 + kgrp * 8;
        a.q[0] = *(const f32x4*)(Ar + off);
        a.q[1] = *(const f32x4*)(Ar + off + 16);
        const f32x4* bs = (const f32x4*)(Bp + (kb * 32 + lane) * 16);
        bb.q[0] = bs[0];
        bb.q[1] = bs[1];
        __builtin_prefetch(Ar + off + 512, 0, 1);   // global_prefetch_b8
        acc = __builtin_amdgcn_wmma_f32_16x16x32_f16(
            false, a.v, false, bb.v, (short)0, acc, false, false);
    }
    const int n = nt * 16 + (lane & 15);
    const float bz = (n < N) ? bias[n] : 0.f;
    #pragma unroll
    for (int r = 0; r < 8; ++r) {
        const int ms = mt16 + r + (kgrp << 3);
        if (ms < M && n < N) {
            float v = acc[r] + bz;
            out[(size_t)ms * N + n] = v > 0.f ? v : 0.f;
        }
    }
}

// ---------------------------------------------------------------------------
// Fused tail: fc2(100->50) fc3(50->10) fc4(10->1) + AEBS MLP (3->12->24->4).
// One thread per batch row; all weights staged in LDS (~24KB).
// ---------------------------------------------------------------------------
__global__ __launch_bounds__(256) void fused_tail(
    const float* __restrict__ x,   // B x 100
    const float* __restrict__ w2, const float* __restrict__ b2,
    const float* __restrict__ w3, const float* __restrict__ b3,
    const float* __restrict__ w4, const float* __restrict__ b4,
    const float* __restrict__ a1w, const float* __restrict__ a1b,
    const float* __restrict__ a2w, const float* __restrict__ a2b,
    const float* __restrict__ a3w, const float* __restrict__ a3b,
    const float* __restrict__ dist, const float* __restrict__ spd,
    float* __restrict__ out, int B)
{
    __shared__ float sW2[5000], sB2[50], sW3[500], sB3[10], sW4[10];
    __shared__ float sA1[36], sA1b[12], sA2[288], sA2b[24], sA3[96], sA3b[4];
    for (int i = threadIdx.x; i < 5000; i += 256) sW2[i] = w2[i];
    for (int i = threadIdx.x; i < 500;  i += 256) sW3[i] = w3[i];
    for (int i = threadIdx.x; i < 288;  i += 256) sA2[i] = a2w[i];
    if (threadIdx.x < 96) sA3[threadIdx.x] = a3w[threadIdx.x];
    if (threadIdx.x < 50) sB2[threadIdx.x] = b2[threadIdx.x];
    if (threadIdx.x < 36) sA1[threadIdx.x] = a1w[threadIdx.x];
    if (threadIdx.x < 24) sA2b[threadIdx.x] = a2b[threadIdx.x];
    if (threadIdx.x < 12) sA1b[threadIdx.x] = a1b[threadIdx.x];
    if (threadIdx.x < 10) { sB3[threadIdx.x] = b3[threadIdx.x]; sW4[threadIdx.x] = w4[threadIdx.x]; }
    if (threadIdx.x < 4)  sA3b[threadIdx.x] = a3b[threadIdx.x];
    __syncthreads();

    const int b = blockIdx.x * 256 + threadIdx.x;
    if (b >= B) return;
    const float* xr = x + (size_t)b * 100;

    float h2[50];
    #pragma unroll
    for (int j = 0; j < 50; ++j) h2[j] = sB2[j];
    for (int i = 0; i < 100; ++i) {
        const float xi = xr[i];
        #pragma unroll
        for (int j = 0; j < 50; ++j) h2[j] += xi * sW2[i * 50 + j];
    }
    #pragma unroll
    for (int j = 0; j < 50; ++j) h2[j] = h2[j] > 0.f ? h2[j] : 0.f;

    float h3[10];
    #pragma unroll
    for (int j = 0; j < 10; ++j) h3[j] = sB3[j];
    #pragma unroll
    for (int i = 0; i < 50; ++i) {
        #pragma unroll
        for (int j = 0; j < 10; ++j) h3[j] += h2[i] * sW3[i * 10 + j];
    }
    #pragma unroll
    for (int j = 0; j < 10; ++j) h3[j] = h3[j] > 0.f ? h3[j] : 0.f;

    float st = b4[0];
    #pragma unroll
    for (int i = 0; i < 10; ++i) st += h3[i] * sW4[i];

    const float s0 = st, s1 = dist[b], s2 = spd[b];
    float ha[12];
    #pragma unroll
    for (int j = 0; j < 12; ++j) {
        float v = sA1b[j] + s0 * sA1[j] + s1 * sA1[12 + j] + s2 * sA1[24 + j];
        ha[j] = v > 0.f ? v : 0.f;
    }
    float hb[24];
    #pragma unroll
    for (int j = 0; j < 24; ++j) hb[j] = sA2b[j];
    #pragma unroll
    for (int i = 0; i < 12; ++i) {
        #pragma unroll
        for (int j = 0; j < 24; ++j) hb[j] += ha[i] * sA2[i * 24 + j];
    }
    #pragma unroll
    for (int j = 0; j < 24; ++j) hb[j] = hb[j] > 0.f ? hb[j] : 0.f;

    #pragma unroll
    for (int j = 0; j < 4; ++j) {
        float o = sA3b[j];
        #pragma unroll
        for (int i = 0; i < 24; ++i) o += hb[i] * sA3[i * 4 + j];
        out[(size_t)b * 4 + j] = o;
    }
}

// ---------------------------------------------------------------------------
static inline unsigned cdiv(long a, long b) { return (unsigned)((a + b - 1) / b); }

extern "C" void kernel_launch(void* const* d_in, const int* in_sizes, int n_in,
                              void* d_out, int out_size, void* d_ws, size_t ws_size,
                              hipStream_t stream) {
    const float* rgb  = (const float*)d_in[0];
    const float* dist = (const float*)d_in[1];
    const float* spd  = (const float*)d_in[2];
    const float* c1w = (const float*)d_in[3];  const float* c1b = (const float*)d_in[4];
    const float* c2w = (const float*)d_in[5];  const float* c2b = (const float*)d_in[6];
    const float* c3w = (const float*)d_in[7];  const float* c3b = (const float*)d_in[8];
    const float* c4w = (const float*)d_in[9];  const float* c4b = (const float*)d_in[10];
    const float* f1w = (const float*)d_in[11]; const float* f1b = (const float*)d_in[12];
    const float* f2w = (const float*)d_in[13]; const float* f2b = (const float*)d_in[14];
    const float* f3w = (const float*)d_in[15]; const float* f3b = (const float*)d_in[16];
    const float* f4w = (const float*)d_in[17]; const float* f4b = (const float*)d_in[18];
    const float* a1w = (const float*)d_in[19]; const float* a1b = (const float*)d_in[20];
    const float* a2w = (const float*)d_in[21]; const float* a2b = (const float*)d_in[22];
    const float* a3w = (const float*)d_in[23]; const float* a3b = (const float*)d_in[24];
    float* outp = (float*)d_out;

    const int B = in_sizes[1];   // batch from distance vector
    const int H0 = 200, W0 = 75;
    const int H1 = 98,  W1 = 36, P1 = H1 * W1;   // conv1 out
    const int H2 = 47,  W2 = 16, P2 = H2 * W2;   // conv2 out
    const int H3 = 22,  W3 = 6,  P3 = H3 * W3;   // conv3 out
    const int H4 = 20,  W4 = 4,  P4 = H4 * W4;   // conv4 out (64*80 = 5120)
    (void)P4; (void)n_in; (void)out_size; (void)ws_size;

    char* ws = (char*)d_ws;
    size_t off = 0;
    auto alloc = [&](size_t n) { size_t r = off; off = (off + n + 255) & ~(size_t)255; return r; };
    _Float16* act0 = (_Float16*)(ws + alloc((size_t)B * 15000 * 2));
    _Float16* act1 = (_Float16*)(ws + alloc((size_t)B * P1 * 24 * 2));
    _Float16* act2 = (_Float16*)(ws + alloc((size_t)B * P2 * 36 * 2));
    _Float16* act3 = (_Float16*)(ws + alloc((size_t)B * P3 * 48 * 2));
    _Float16* act4 = (_Float16*)(ws + alloc((size_t)B * 5120 * 2));    // NCHW flat
    float*    fc1o = (float*)   (ws + alloc((size_t)B * 100 * 4));
    _Float16* wp1  = (_Float16*)(ws + alloc((size_t)2 * 1  * 512 * 2));
    _Float16* wp2  = (_Float16*)(ws + alloc((size_t)3 * 19 * 512 * 2));
    _Float16* wp3  = (_Float16*)(ws + alloc((size_t)3 * 29 * 512 * 2));
    _Float16* wp4  = (_Float16*)(ws + alloc((size_t)4 * 14 * 512 * 2));
    _Float16* wpf  = (_Float16*)(ws + alloc((size_t)7 * 160 * 512 * 2));

    binarize_transpose<<<cdiv((long)B * 15000, 256), 256, 0, stream>>>(rgb, act0, B);

    pack_conv_w<1, 24, 5><<<cdiv(2 * 1 * 512, 256), 256, 0, stream>>>(c1w, wp1);
    pack_conv_w<24, 36, 5><<<cdiv(3 * 19 * 512, 256), 256, 0, stream>>>(c2w, wp2);
    pack_conv_w<36, 48, 5><<<cdiv(3 * 29 * 512, 256), 256, 0, stream>>>(c3w, wp3);
    pack_conv_w<48, 64, 3><<<cdiv(4 * 14 * 512, 256), 256, 0, stream>>>(c4w, wp4);
    pack_fc_w<<<cdiv(7 * 160 * 512, 256), 256, 0, stream>>>(f1w, wpf, 5120, 100);

    // dynamic LDS bytes = NT*KC*1024
    conv_wmma<1, 24, 5, 2, false><<<cdiv((long)B * P1, 128), 256, 2 * 1 * 1024, stream>>>(
        act0, wp1, c1b, act1, H0, W0, H1, W1, B);
    conv_wmma<24, 36, 5, 2, false><<<cdiv((long)B * P2, 128), 256, 3 * 19 * 1024, stream>>>(
        act1, wp2, c2b, act2, H1, W1, H2, W2, B);
    conv_wmma<36, 48, 5, 2, false><<<cdiv((long)B * P3, 128), 256, 3 * 29 * 1024, stream>>>(
        act2, wp3, c3b, act3, H2, W2, H3, W3, B);
    conv_wmma<48, 64, 3, 1, true><<<cdiv((long)B * 80, 128), 256, 4 * 14 * 1024, stream>>>(
        act3, wp4, c4b, act4, H3, W3, H4, W4, B);

    gemm_wmma_f16<<<dim3(cdiv(B, 128), 7), 256, 0, stream>>>(act4, wpf, f1b, fc1o, B, 5120, 100);

    fused_tail<<<cdiv(B, 256), 256, 0, stream>>>(
        fc1o, f2w, f2b, f3w, f3b, f4w, f4b,
        a1w, a1b, a2w, a2b, a3w, a3b, dist, spd, outp, B);
}
